// Attention_17652315587089
// MI455X (gfx1250) — matrix operand order repaired and checked by reading, not compiled, not run on hardware
//
#include <hip/hip_runtime.h>
#include <hip/hip_bf16.h>
#include <stdint.h>

// ---------------------------------------------------------------------------
// Types for CDNA5 WMMA (wave32, 16x16x32 bf16 -> f32)
// ---------------------------------------------------------------------------
typedef __bf16    v16bf __attribute__((ext_vector_type(16)));
typedef float     v8f   __attribute__((ext_vector_type(8)));
typedef uint32_t  u32x4 __attribute__((ext_vector_type(4)));
typedef int       i32x4 __attribute__((ext_vector_type(4)));
typedef int       i32x8 __attribute__((ext_vector_type(8)));

static constexpr int B_  = 2;
static constexpr int T_  = 2048;
static constexpr int D_  = 1024;
static constexpr int H_  = 16;
static constexpr int E_  = 3 * D_;     // 3072
static constexpr int BT_ = B_ * T_;    // 4096
static constexpr int TD_ = T_ * D_;    // 2097152 (power of two)

__device__ __forceinline__ unsigned short f2bf(float f) {
    union { float f; uint32_t u; } c; c.f = f;
    uint32_t u = c.u;
    u += 0x7fffu + ((u >> 16) & 1u);          // round-to-nearest-even
    return (unsigned short)(u >> 16);
}

__device__ __forceinline__ v8f zero8() {
    v8f z;
#pragma unroll
    for (int i = 0; i < 8; ++i) z[i] = 0.0f;
    return z;
}

__device__ __forceinline__ v8f wmma_bf16(v16bf a, v16bf b, v8f c) {
    return __builtin_amdgcn_wmma_f32_16x16x32_bf16(
        /*neg_a=*/false, a, /*neg_b=*/false, b,
        /*c_mod=*/(short)0, c, /*reuse_a=*/false, /*reuse_b=*/false);
}

// A fragment: 16x32 bf16 tile from row-major src (element (m,k) at src[m*stride+k]).
// Lane L: row = L&15, kb = (L>>4)*8; holds K = {kb..kb+7, kb+16..kb+23} -> two b128.
__device__ __forceinline__ v16bf load_a_frag(const unsigned short* src, int stride,
                                             int row0, int k0, int lane) {
    const int m  = lane & 15;
    const int kb = (lane >> 4) * 8;
    union { v16bf v; u32x4 q[2]; } f;
    const unsigned short* p = src + (row0 + m) * stride + k0 + kb;
    f.q[0] = *reinterpret_cast<const u32x4*>(p);
    f.q[1] = *reinterpret_cast<const u32x4*>(p + 16);
    return f.v;
}

// B fragment: 32x16 bf16 tile from N-major src (element (k,n) at src[n*stride+k]).
// Lane L: col = L&15, kb = (L>>4)*8; holds K = {kb..kb+7, kb+16..kb+23} -> two b128.
__device__ __forceinline__ v16bf load_b_frag(const unsigned short* src, int stride,
                                             int n0, int k0, int lane) {
    const int n  = lane & 15;
    const int kb = (lane >> 4) * 8;
    union { v16bf v; u32x4 q[2]; } f;
    const unsigned short* p = src + (n0 + n) * stride + k0 + kb;
    f.q[0] = *reinterpret_cast<const u32x4*>(p);
    f.q[1] = *reinterpret_cast<const u32x4*>(p + 16);
    return f.v;
}

// ---------------------------------------------------------------------------
// Kernel 1a: x = query + pos_table, cast to bf16
// ---------------------------------------------------------------------------
__global__ void k_xconv(const float* __restrict__ q, const float* __restrict__ pos,
                        unsigned short* __restrict__ xb) {
    const int i = blockIdx.x * blockDim.x + threadIdx.x;
    if (i < B_ * T_ * D_) xb[i] = f2bf(q[i] + pos[i & (TD_ - 1)]);
}

// Kernel 1b: generic fp32 -> bf16 cast
__global__ void k_conv(const float* __restrict__ s, unsigned short* __restrict__ d, int n) {
    const int i = blockIdx.x * blockDim.x + threadIdx.x;
    if (i < n) d[i] = f2bf(s[i]);
}

// ---------------------------------------------------------------------------
// Kernel 2: qkv = x @ Wqkv^T, software-pipelined, 32(row) x 64(col) per wave.
// Epilogue scatters q,k row-major [b*T+t][D]; v transposed vT[b][h][dh][T].
// ---------------------------------------------------------------------------
__global__ void k_qkv(const unsigned short* __restrict__ xb,
                      const unsigned short* __restrict__ wb,
                      unsigned short* __restrict__ qb,
                      unsigned short* __restrict__ kb,
                      unsigned short* __restrict__ vtb) {
    const int lane = threadIdx.x;
    const int row0 = blockIdx.x * 32;
    const int e0   = blockIdx.y * 64;

    v8f acc[2][4];
#pragma unroll
    for (int rb = 0; rb < 2; ++rb)
#pragma unroll
        for (int t = 0; t < 4; ++t) acc[rb][t] = zero8();

    // Prologue: fragments for k0 = 0
    v16bf a_cur[2], b_cur[4];
#pragma unroll
    for (int rb = 0; rb < 2; ++rb) a_cur[rb] = load_a_frag(xb, D_, row0 + 16 * rb, 0, lane);
#pragma unroll
    for (int t = 0; t < 4; ++t)    b_cur[t]  = load_b_frag(wb, D_, e0 + 16 * t, 0, lane);

    for (int k0 = 0; k0 < D_; k0 += 32) {
        const int kn = (k0 + 32) & (D_ - 1);       // branchless wrap (dead on last iter)
        v16bf a_nxt[2], b_nxt[4];
#pragma unroll
        for (int rb = 0; rb < 2; ++rb) a_nxt[rb] = load_a_frag(xb, D_, row0 + 16 * rb, kn, lane);
#pragma unroll
        for (int t = 0; t < 4; ++t)    b_nxt[t]  = load_b_frag(wb, D_, e0 + 16 * t, kn, lane);
        __builtin_prefetch(xb + (row0 + (lane & 15)) * D_ + kn + 32, 0, 3);

#pragma unroll
        for (int rb = 0; rb < 2; ++rb)
#pragma unroll
            for (int t = 0; t < 4; ++t)
                acc[rb][t] = wmma_bf16(a_cur[rb], b_cur[t], acc[rb][t]);

#pragma unroll
        for (int rb = 0; rb < 2; ++rb) a_cur[rb] = a_nxt[rb];
#pragma unroll
        for (int t = 0; t < 4; ++t)    b_cur[t]  = b_nxt[t];
    }

    const int n = lane & 15, mo = (lane >> 4) * 8;
#pragma unroll
    for (int rb = 0; rb < 2; ++rb) {
#pragma unroll
        for (int t = 0; t < 4; ++t) {
            const int e = e0 + 16 * t + n;
            const int c = e >> 10;          // 0=q, 1=k, 2=v
            const int w = e & 1023;         // h*64 + d
#pragma unroll
            for (int r = 0; r < 8; ++r) {
                const int g = row0 + 16 * rb + mo + r;   // global row = b*T + t
                const unsigned short bv = f2bf(acc[rb][t][r]);
                if (c == 0)      qb[g * D_ + w] = bv;
                else if (c == 1) kb[g * D_ + w] = bv;
                else {
                    const int bi = g >> 11, ti = g & (T_ - 1);
                    const int h = w >> 6, d = w & 63;
                    vtb[(((bi * H_ + h) << 6) + d) * T_ + ti] = bv;
                }
            }
        }
    }
}

// ---------------------------------------------------------------------------
// Kernel 3: flash attention, one wave per (b, h, 16-query tile), key tiles of 32.
// Next k-tile and current v-tile fragments are issued before the softmax phase
// so VMEM latency hides under exp/LDS work. A TDM tensor_load_to_lds stages the
// first key tile into LDS (gfx1250 async-tensor path; compute keeps the
// global-load route).
// ---------------------------------------------------------------------------
__global__ void k_attn(const unsigned short* __restrict__ qb,
                       const unsigned short* __restrict__ kb,
                       const unsigned short* __restrict__ vtb,
                       const unsigned char* __restrict__ qmask,
                       unsigned short* __restrict__ yb) {
    const int lane = threadIdx.x;
    const int q0 = blockIdx.x * 16;
    const int h  = blockIdx.y;
    const int b  = blockIdx.z;

    const unsigned short* qsrc = qb  + (b * T_) * D_ + h * 64;       // (t,d) at t*D+d
    const unsigned short* ksrc = kb  + (b * T_) * D_ + h * 64;       // (j,d) at j*D+d
    const unsigned short* vsrc = vtb + ((b * H_ + h) << 6) * T_;     // (d,j) at d*T+j
    const unsigned char*  mrow = qmask + b * T_;

    __shared__ alignas(16) float          Sbuf[16 * 32];
    __shared__ alignas(16) unsigned short Pbuf[16 * 32];
    __shared__ float Abuf[16];
    __shared__ float Lbuf[16];

#if __has_builtin(__builtin_amdgcn_tensor_load_to_lds)
    // --- Tensor Data Mover: DMA the first 32(keys) x 64(dh) bf16 k-tile into a
    // dedicated LDS staging buffer (D# per CDNA5 ISA 8.3/8.4; count=1, type=2,
    // data_size=2B, tile 64x32, dim0_stride=D). Acts as an LDS/L2 warm prefetch.
    __shared__ alignas(16) unsigned short TdmStage[32 * 64];
    {
        const uint64_t ga  = (uint64_t)(uintptr_t)ksrc;              // byte address
        const uint32_t lds = (uint32_t)(uintptr_t)(&TdmStage[0]);    // LDS byte offset
        const u32x4 g0 = { 1u,                                       // count=1
                           lds,                                      // lds_addr
                           (uint32_t)ga,                             // global_addr[31:0]
                           (uint32_t)((ga >> 32) & 0x01FFFFFFu) | (2u << 30) }; // type=2
        const i32x8 g1 = { (int)(1u << 16),          // data_size=1 (2 bytes)
                           (int)(64u << 16),         // tensor_dim0 = 64
                           (int)(32u << 16),         // tensor_dim1 = 32
                           (int)(64u << 16),         // tile_dim0   = 64
                           (int)32,                  // tile_dim1   = 32
                           (int)D_,                  // tensor_dim0_stride = 1024
                           0, 0 };
        const i32x4 gz4 = { 0, 0, 0, 0 };
        const i32x8 gz8 = { 0, 0, 0, 0, 0, 0, 0, 0 };
        // 6-arg (clang-23 / therock-10.0) form: groups 0-3 + extra group + cpol
        __builtin_amdgcn_tensor_load_to_lds(g0, g1, gz4, gz4, gz8, 0);
    }
#endif

    const v16bf qa0 = load_a_frag(qsrc, D_, q0, 0,  lane);
    const v16bf qa1 = load_a_frag(qsrc, D_, q0, 32, lane);

    v8f O[4];
#pragma unroll
    for (int t = 0; t < 4; ++t) O[t] = zero8();
    float m_run = -3e38f, l_run = 0.0f;

    const int n = lane & 15, mo = (lane >> 4) * 8;
    const int ntiles = (q0 + 16 + 31) >> 5;                // causal: keys < q0+16

    // Prologue: k fragments for tile 0
    v16bf kf[4];
    kf[0] = load_b_frag(ksrc, D_, 0,  0,  lane);
    kf[1] = load_b_frag(ksrc, D_, 0,  32, lane);
    kf[2] = load_b_frag(ksrc, D_, 16, 0,  lane);
    kf[3] = load_b_frag(ksrc, D_, 16, 32, lane);

    for (int it = 0; it < ntiles; ++it) {
        const int j0 = it << 5;

        // S = (q @ k^T): two 16-key N-subtiles, K=64 split into 32+32
        v8f S0 = zero8(), S1 = zero8();
        S0 = wmma_bf16(qa0, kf[0], S0);
        S0 = wmma_bf16(qa1, kf[1], S0);
        S1 = wmma_bf16(qa0, kf[2], S1);
        S1 = wmma_bf16(qa1, kf[3], S1);

        // Issue next k-tile + current v-tile loads now; they retire during softmax.
        const int itn = (it + 1 < ntiles) ? it + 1 : it;
        const int jn  = itn << 5;
        v16bf kn[4], vf[4];
        kn[0] = load_b_frag(ksrc, D_, jn,      0,  lane);
        kn[1] = load_b_frag(ksrc, D_, jn,      32, lane);
        kn[2] = load_b_frag(ksrc, D_, jn + 16, 0,  lane);
        kn[3] = load_b_frag(ksrc, D_, jn + 16, 32, lane);
#pragma unroll
        for (int t = 0; t < 4; ++t) vf[t] = load_b_frag(vsrc, T_, t * 16, j0, lane);

        // Stage scaled + masked scores into LDS (C-layout -> row-major 16x32)
        const int jA = j0 + n, jB = j0 + 16 + n;
        const bool padA = mrow[jA] != 0, padB = mrow[jB] != 0;
#pragma unroll
        for (int r = 0; r < 8; ++r) {
            const int qrow = q0 + mo + r;
            Sbuf[(mo + r) * 32 + n]      = (padA || jA > qrow) ? -1e30f : S0[r] * 0.125f;
            Sbuf[(mo + r) * 32 + 16 + n] = (padB || jB > qrow) ? -1e30f : S1[r] * 0.125f;
        }
        __syncthreads();   // single-wave WG: barrier is S_NOP, acts as fence

        // Online softmax: lanes 0..15 each own one query row
        if (lane < 16) {
            const float* sr = Sbuf + lane * 32;
            float tm = -3e38f;
#pragma unroll
            for (int jj = 0; jj < 32; ++jj) tm = fmaxf(tm, sr[jj]);
            const float m_new = fmaxf(m_run, tm);
            const float alpha = __expf(m_run - m_new);
            float ssum = 0.0f;
#pragma unroll
            for (int jj = 0; jj < 32; ++jj) {
                const float sv = sr[jj];
                const float p = (sv <= -1e29f) ? 0.0f : __expf(sv - m_new);
                ssum += p;
                Pbuf[lane * 32 + jj] = f2bf(p);
            }
            l_run = l_run * alpha + ssum;
            m_run = m_new;
            Abuf[lane] = alpha;
        }
        __syncthreads();

        // Rescale O accumulators by per-row alpha
        float al[8];
#pragma unroll
        for (int r = 0; r < 8; ++r) al[r] = Abuf[mo + r];
#pragma unroll
        for (int t = 0; t < 4; ++t)
#pragma unroll
            for (int r = 0; r < 8; ++r) O[t][r] *= al[r];

        // O += P(16x32) @ V(32x64): P is exactly one A fragment
        const v16bf pa = load_a_frag(Pbuf, 32, 0, 0, lane);
#pragma unroll
        for (int t = 0; t < 4; ++t)
            O[t] = wmma_bf16(pa, vf[t], O[t]);

#pragma unroll
        for (int t = 0; t < 4; ++t) kf[t] = kn[t];
        __syncthreads();
    }

#if __has_builtin(__builtin_amdgcn_s_wait_tensorcnt)
    __builtin_amdgcn_s_wait_tensorcnt(0);   // drain TDM before wave end
#endif

    // Finalize: divide by row sum, zero masked query rows, write y bf16
    if (lane < 16) {
        Lbuf[lane] = (mrow[q0 + lane] != 0) ? 0.0f : 1.0f / fmaxf(l_run, 1e-30f);
    }
    __syncthreads();
    float linv[8];
#pragma unroll
    for (int r = 0; r < 8; ++r) linv[r] = Lbuf[mo + r];
#pragma unroll
    for (int t = 0; t < 4; ++t)
#pragma unroll
        for (int r = 0; r < 8; ++r)
            yb[((b * T_) + q0 + mo + r) * D_ + h * 64 + t * 16 + n] = f2bf(O[t][r] * linv[r]);
}

// ---------------------------------------------------------------------------
// Kernel 4: out = y @ Wo^T, software-pipelined, 32x64 per wave, fp32 output
// ---------------------------------------------------------------------------
__global__ void k_out(const unsigned short* __restrict__ yb,
                      const unsigned short* __restrict__ wob,
                      float* __restrict__ out) {
    const int lane = threadIdx.x;
    const int row0 = blockIdx.x * 32;
    const int c0   = blockIdx.y * 64;

    v8f acc[2][4];
#pragma unroll
    for (int rb = 0; rb < 2; ++rb)
#pragma unroll
        for (int t = 0; t < 4; ++t) acc[rb][t] = zero8();

    v16bf a_cur[2], b_cur[4];
#pragma unroll
    for (int rb = 0; rb < 2; ++rb) a_cur[rb] = load_a_frag(yb, D_, row0 + 16 * rb, 0, lane);
#pragma unroll
    for (int t = 0; t < 4; ++t)    b_cur[t]  = load_b_frag(wob, D_, c0 + 16 * t, 0, lane);

    for (int k0 = 0; k0 < D_; k0 += 32) {
        const int kn = (k0 + 32) & (D_ - 1);
        v16bf a_nxt[2], b_nxt[4];
#pragma unroll
        for (int rb = 0; rb < 2; ++rb) a_nxt[rb] = load_a_frag(yb, D_, row0 + 16 * rb, kn, lane);
#pragma unroll
        for (int t = 0; t < 4; ++t)    b_nxt[t]  = load_b_frag(wob, D_, c0 + 16 * t, kn, lane);
        __builtin_prefetch(yb + (row0 + (lane & 15)) * D_ + kn + 32, 0, 3);

#pragma unroll
        for (int rb = 0; rb < 2; ++rb)
#pragma unroll
            for (int t = 0; t < 4; ++t)
                acc[rb][t] = wmma_bf16(a_cur[rb], b_cur[t], acc[rb][t]);

#pragma unroll
        for (int rb = 0; rb < 2; ++rb) a_cur[rb] = a_nxt[rb];
#pragma unroll
        for (int t = 0; t < 4; ++t)    b_cur[t]  = b_nxt[t];
    }

    const int n = lane & 15, mo = (lane >> 4) * 8;
#pragma unroll
    for (int rb = 0; rb < 2; ++rb)
#pragma unroll
        for (int t = 0; t < 4; ++t)
#pragma unroll
            for (int r = 0; r < 8; ++r)
                out[(row0 + 16 * rb + mo + r) * D_ + c0 + 16 * t + n] = acc[rb][t][r];
}

// ---------------------------------------------------------------------------
// Host-side launch (graph-capture safe; scratch partitioned from d_ws)
// ---------------------------------------------------------------------------
extern "C" void kernel_launch(void* const* d_in, const int* in_sizes, int n_in,
                              void* d_out, int out_size, void* d_ws, size_t ws_size,
                              hipStream_t stream) {
    (void)in_sizes; (void)n_in; (void)out_size; (void)ws_size;

    const float*         query = (const float*)d_in[0];
    const unsigned char* qmask = (const unsigned char*)d_in[1];
    const float*         Wqkv  = (const float*)d_in[2];
    const float*         Wo    = (const float*)d_in[3];
    const float*         pos   = (const float*)d_in[4];
    float*               out   = (float*)d_out;

    // Workspace layout (bf16 buffers, 48 MB total; all L2-resident on MI455X)
    uint8_t* ws = (uint8_t*)d_ws;
    unsigned short* xb    = (unsigned short*)(ws + (size_t) 0 * 1024 * 1024); //  8 MB
    unsigned short* wqkvb = (unsigned short*)(ws + (size_t) 8 * 1024 * 1024); //  6 MB
    unsigned short* wob   = (unsigned short*)(ws + (size_t)14 * 1024 * 1024); //  2 MB
    unsigned short* qbuf  = (unsigned short*)(ws + (size_t)16 * 1024 * 1024); //  8 MB
    unsigned short* kbuf  = (unsigned short*)(ws + (size_t)24 * 1024 * 1024); //  8 MB
    unsigned short* vtb   = (unsigned short*)(ws + (size_t)32 * 1024 * 1024); //  8 MB
    unsigned short* ybuf  = (unsigned short*)(ws + (size_t)40 * 1024 * 1024); //  8 MB

    k_xconv<<<(B_ * T_ * D_ + 255) / 256, 256, 0, stream>>>(query, pos, xb);
    k_conv <<<(3 * D_ * D_ + 255) / 256, 256, 0, stream>>>(Wqkv, wqkvb, 3 * D_ * D_);
    k_conv <<<(D_ * D_ + 255) / 256,     256, 0, stream>>>(Wo, wob, D_ * D_);

    k_qkv <<<dim3(BT_ / 32, E_ / 64), 32, 0, stream>>>(xb, wqkvb, qbuf, kbuf, vtb);
    k_attn<<<dim3(T_ / 16, H_, B_),   32, 0, stream>>>(qbuf, kbuf, vtb, qmask, ybuf);
    k_out <<<dim3(BT_ / 32, D_ / 64), 32, 0, stream>>>(ybuf, wob, out);
}
